// GAT_64561948393643
// MI455X (gfx1250) — compile-verified, matrix operand order checked by reading, not run on hardware
//
#include <hip/hip_runtime.h>
#include <hip/hip_bf16.h>
#include <cstdint>
#include <cstddef>

// ---------------------------------------------------------------------------
// CDNA5 / gfx1250: wave32, WMMA 16x16x32 bf16 for the dense node projections.
// LDS tiles are kept in WMMA *fragment order* so fragment loads are single
// 32B ds reads and staging stores are b128 ds (A global loads are b128, B's are scalar).
// ---------------------------------------------------------------------------

typedef __attribute__((ext_vector_type(16))) __bf16 v16bf;
typedef __attribute__((ext_vector_type(8)))  __bf16 v8bf;
typedef __attribute__((ext_vector_type(8)))  float  v8f;

#define NNODES 100000
#define NEDGES 1600000
#define NGRAPH 256

__device__ __forceinline__ float leakyf(float x, float s) {
    return x >= 0.f ? x : s * x;
}

__device__ __forceinline__ void atomicMaxF(float* addr, float val) {
    unsigned int* ua = (unsigned int*)addr;
    unsigned int old = *ua, assumed;
    while (__uint_as_float(old) < val) {
        assumed = old;
        old = atomicCAS(ua, assumed, __float_as_uint(val));
        if (old == assumed) break;
    }
}

// ---------------------------------------------------------------------------
// WMMA GEMM:  C[M x 64] = A[M x K](f32) @ W[K x 64](f32), bf16 compute.
// Block = 256 threads (8 waves); 128-row x 64-col tile per block.
// Fragment-order LDS layout:
//   A lane/slot map (16x32 bf16): lane = (row%16) + 16*((k>>3)&1),
//                                 slot = (k&7) + 8*(k>>4)
//   B lane/slot map (32x16 bf16): lane = (col%16) + 16*((k>>3)&1), same slot.
// Software pipelined: next tile's global loads issue while WMMAs consume LDS.
// ---------------------------------------------------------------------------
template <bool FUSE_EMB>
__global__ __launch_bounds__(256)
void k_gemm64_wmma(const float* __restrict__ A, int lda,
                   const float* __restrict__ Wm, int ldw,
                   float* __restrict__ C, int M, int K,
                   const float* __restrict__ Wfull,
                   const float* __restrict__ emb,
                   const float* __restrict__ emb2,
                   const int* __restrict__ res,
                   const int* __restrict__ sec) {
    constexpr int KT = 32;
    __shared__ __align__(32) __bf16 As[8][32][16];   // 8 KB, fragment order
    __shared__ __align__(32) __bf16 Bs[4][32][16];   // 4 KB, fragment order

    const int tid  = threadIdx.x;
    const int wave = tid >> 5;
    const int lane = tid & 31;
    const int rowBase = blockIdx.x * 128;
    const int lr = lane & 15;

    // --- staging thread roles
    const int aRow   = tid >> 1;          // 0..127
    const int aHalf  = tid & 1;           // k-half: 0 -> k0..15, 1 -> k16..31
    const int aStrip = aRow >> 4;
    const int aLane  = aRow & 15;
    const int aSlot  = aHalf * 8;
    const int bCol   = tid & 63;          // 0..63
    const int bKq    = tid >> 6;          // 0..3 (8 consecutive k each)
    const int bTile  = bCol >> 4;
    const int bLane  = (bCol & 15) + 16 * (bKq & 1);
    const int bSlot  = (bKq >> 1) * 8;

    float4 aR[4];
    float  bR[8];

    auto loadTiles = [&](int kt) {
        int gr = rowBase + aRow;
        if (gr < M) {
            const float* p = A + (size_t)gr * lda + kt + aHalf * 16;
            aR[0] = *(const float4*)(p + 0);
            aR[1] = *(const float4*)(p + 4);
            aR[2] = *(const float4*)(p + 8);
            aR[3] = *(const float4*)(p + 12);
        } else {
            aR[0] = aR[1] = aR[2] = aR[3] = make_float4(0.f, 0.f, 0.f, 0.f);
        }
        const float* q = Wm + (size_t)(kt + bKq * 8) * ldw + bCol;
        #pragma unroll
        for (int j = 0; j < 8; ++j) bR[j] = q[(size_t)j * ldw];
    };
    auto storeTiles = [&]() {
        v8bf lo, hi;
        lo[0] = (__bf16)aR[0].x; lo[1] = (__bf16)aR[0].y;
        lo[2] = (__bf16)aR[0].z; lo[3] = (__bf16)aR[0].w;
        lo[4] = (__bf16)aR[1].x; lo[5] = (__bf16)aR[1].y;
        lo[6] = (__bf16)aR[1].z; lo[7] = (__bf16)aR[1].w;
        hi[0] = (__bf16)aR[2].x; hi[1] = (__bf16)aR[2].y;
        hi[2] = (__bf16)aR[2].z; hi[3] = (__bf16)aR[2].w;
        hi[4] = (__bf16)aR[3].x; hi[5] = (__bf16)aR[3].y;
        hi[6] = (__bf16)aR[3].z; hi[7] = (__bf16)aR[3].w;
        *(v8bf*)&As[aStrip][aLane][aSlot]      = lo;  // k .. k+7   (lane lo)
        *(v8bf*)&As[aStrip][aLane + 16][aSlot] = hi;  // k+8 .. k+15 (lane hi)
        v8bf bb;
        #pragma unroll
        for (int j = 0; j < 8; ++j) bb[j] = (__bf16)bR[j];
        *(v8bf*)&Bs[bTile][bLane][bSlot] = bb;
    };

    v8f acc[4] = {};

    loadTiles(0);
    for (int kt = 0; kt < K; kt += KT) {
        storeTiles();
        __syncthreads();
        if (kt + KT < K) loadTiles(kt + KT);   // overlap next HBM fetch with WMMA

        v16bf af = *(const v16bf*)&As[wave][lane][0];
        #pragma unroll
        for (int t = 0; t < 4; ++t) {
            v16bf bf = *(const v16bf*)&Bs[t][lane][0];
            acc[t] = __builtin_amdgcn_wmma_f32_16x16x32_bf16(
                false, af, false, bf, (short)0, acc[t], false, false);
        }
        __syncthreads();
    }

    // Fused embedding-concat contribution (columns 0..32 of the 1057-wide input)
    if (FUSE_EMB) {
        #pragma unroll
        for (int t = 0; t < 4; ++t) {
            int col = t * 16 + lr;
            #pragma unroll
            for (int j = 0; j < 8; ++j) {
                int gr = rowBase + wave * 16 + (lane >> 4) * 8 + j;
                if (gr < M) {
                    int ri = res[gr], si = sec[gr];
                    float s = 0.f;
                    #pragma unroll
                    for (int k = 0; k < 21; ++k)
                        s += emb[ri * 21 + k] * Wfull[k * 64 + col];
                    #pragma unroll
                    for (int k = 0; k < 12; ++k)
                        s += emb2[si * 12 + k] * Wfull[(21 + k) * 64 + col];
                    acc[t][j] += s;
                }
            }
        }
    }

    // store: C layout VGPR j -> row 8*(lane>>4)+j, col = lane&15 (+16t)
    #pragma unroll
    for (int t = 0; t < 4; ++t) {
        int col = t * 16 + lr;
        #pragma unroll
        for (int j = 0; j < 8; ++j) {
            int gr = rowBase + wave * 16 + (lane >> 4) * 8 + j;
            if (gr < M) C[(size_t)gr * 64 + col] = acc[t][j];
        }
    }
}

// last layer: [N x 64] @ [64 x 4] — tiny, scalar
__global__ void k_gemm_small(const float* __restrict__ A,
                             const float* __restrict__ Wm,
                             float* __restrict__ C, int M, int K, int Cn) {
    int n = blockIdx.x * blockDim.x + threadIdx.x;
    if (n >= M) return;
    for (int c = 0; c < Cn; ++c) {
        float s = 0.f;
        for (int k = 0; k < K; ++k) s += A[(size_t)n * K + k] * Wm[k * Cn + c];
        C[(size_t)n * Cn + c] = s;
    }
}

// fe = e @ W_e ;  ee[h] = sum_d fe*attn_e
__global__ void k_edge_fe_ee(const float* __restrict__ ein,
                             const float* __restrict__ We,
                             const float* __restrict__ ae,
                             float* __restrict__ fe, float* __restrict__ ee,
                             int E, int Cin, int H, int de) {
    int e = blockIdx.x * blockDim.x + threadIdx.x;
    if (e >= E) return;
    int HD = H * de;
    for (int h = 0; h < H; ++h) {
        float see = 0.f;
        for (int d = 0; d < de; ++d) {
            float s = 0.f;
            for (int k = 0; k < Cin; ++k)
                s += ein[(size_t)e * Cin + k] * We[k * HD + h * de + d];
            fe[(size_t)e * HD + h * de + d] = s;
            see += s * ae[h * de + d];
        }
        ee[(size_t)e * H + h] = see;
    }
}

__global__ void k_node_el_er(const float* __restrict__ ft,
                             const float* __restrict__ al,
                             const float* __restrict__ ar,
                             float* __restrict__ el, float* __restrict__ er,
                             int N, int H, int dn) {
    int n = blockIdx.x * blockDim.x + threadIdx.x;
    if (n >= N) return;
    int HD = H * dn;
    for (int h = 0; h < H; ++h) {
        float sl = 0.f, sr = 0.f;
        for (int d = 0; d < dn; ++d) {
            float f = ft[(size_t)n * HD + h * dn + d];
            sl += f * al[h * dn + d];
            sr += f * ar[h * dn + d];
        }
        el[(size_t)n * H + h] = sl;
        er[(size_t)n * H + h] = sr;
    }
}

__global__ void k_edge_max(const float* __restrict__ el, const float* __restrict__ er,
                           const float* __restrict__ ee, const int* __restrict__ src,
                           const int* __restrict__ dst, float* __restrict__ mb,
                           long EH, int H) {
    long i = (long)blockIdx.x * blockDim.x + threadIdx.x;
    if (i >= EH) return;
    long e = i / H; int h = (int)(i - e * H);
    int s = src[e], d = dst[e];
    float lg = leakyf(el[(size_t)s * H + h] + er[(size_t)d * H + h] + ee[(size_t)e * H + h], 0.2f);
    atomicMaxF(&mb[(size_t)d * H + h], lg);
}

__global__ void k_edge_expsum(const float* __restrict__ el, const float* __restrict__ er,
                              const float* __restrict__ ee, const int* __restrict__ src,
                              const int* __restrict__ dst, const float* __restrict__ mb,
                              float* __restrict__ dnm, float* __restrict__ exb,
                              long EH, int H) {
    long i = (long)blockIdx.x * blockDim.x + threadIdx.x;
    if (i >= EH) return;
    long e = i / H; int h = (int)(i - e * H);
    int s = src[e], d = dst[e];
    float lg = leakyf(el[(size_t)s * H + h] + er[(size_t)d * H + h] + ee[(size_t)e * H + h], 0.2f);
    float mm = mb[(size_t)d * H + h];
    if (mm < -1e29f) mm = 0.f;          // isfinite fixup (no-incoming-edge nodes)
    float ex = __expf(lg - mm);
    exb[i] = ex;
    atomicAdd(&dnm[(size_t)d * H + h], ex);
}

__global__ void k_edge_scatter(const float* __restrict__ ft, const float* __restrict__ exb,
                               const float* __restrict__ dnm, const int* __restrict__ src,
                               const int* __restrict__ dst, float* __restrict__ outb,
                               long EH, int H, int dn) {
    long i = (long)blockIdx.x * blockDim.x + threadIdx.x;
    if (i >= EH) return;
    long e = i / H; int h = (int)(i - e * H);
    int s = src[e], d = dst[e];
    float a = exb[i] / (dnm[(size_t)d * H + h] + 1e-9f);
    const float* fs = &ft[(size_t)s * H * dn + h * dn];
    float* od = &outb[(size_t)d * H * dn + h * dn];
    for (int k = 0; k < dn; ++k) atomicAdd(&od[k], fs[k] * a);
}

__global__ void k_fill(float* __restrict__ p, long n, float v) {
    long i = (long)blockIdx.x * blockDim.x + threadIdx.x;
    if (i < n) p[i] = v;
}

__global__ void k_add(float* __restrict__ x, const float* __restrict__ y, long n) {
    long i = (long)blockIdx.x * blockDim.x + threadIdx.x;
    if (i < n) x[i] += y[i];
}

// per-column sum & sumsq via LDS float atomics (ds_add_f32) + one global atomic
__global__ void k_colstats(const float* __restrict__ X, long Nrows, int C,
                           float* __restrict__ sums) {
    extern __shared__ float sm[];
    for (int i = threadIdx.x; i < 2 * C; i += blockDim.x) sm[i] = 0.f;
    __syncthreads();
    for (long r = (long)blockIdx.x * blockDim.x + threadIdx.x; r < Nrows;
         r += (long)gridDim.x * blockDim.x) {
        for (int c = 0; c < C; ++c) {
            float v = X[(size_t)r * C + c];
            atomicAdd(&sm[c], v);
            atomicAdd(&sm[C + c], v * v);
        }
    }
    __syncthreads();
    for (int i = threadIdx.x; i < 2 * C; i += blockDim.x) atomicAdd(&sums[i], sm[i]);
}

__global__ void k_bn_apply(float* __restrict__ X, long Nrows, int C,
                           const float* __restrict__ sums,
                           const float* __restrict__ g, const float* __restrict__ b,
                           float slope) {
    long i = (long)blockIdx.x * blockDim.x + threadIdx.x;
    if (i >= Nrows * C) return;
    int c = (int)(i % C);
    float invN = 1.f / (float)Nrows;
    float mu = sums[c] * invN;
    float var = sums[C + c] * invN - mu * mu;
    float y = (X[i] - mu) * rsqrtf(var + 1e-5f) * g[c] + b[c];
    if (slope >= 0.f) y = leakyf(y, slope);
    X[i] = y;
}

__global__ void k_seg_sum(const float* __restrict__ x, const int* __restrict__ gids,
                          float* __restrict__ hg, int N) {
    int n = blockIdx.x * blockDim.x + threadIdx.x;
    if (n >= N) return;
    int g = gids[n];
    for (int c = 0; c < 4; ++c) atomicAdd(&hg[g * 4 + c], x[(size_t)n * 4 + c]);
}

__global__ void k_final(const float* __restrict__ hg, const float* __restrict__ Wa,
                        const float* __restrict__ ba, float* __restrict__ out) {
    int g = blockIdx.x * blockDim.x + threadIdx.x;
    if (g >= NGRAPH) return;
    float v[4], l[4];
    for (int c = 0; c < 4; ++c) { v[c] = hg[g * 4 + c]; l[c] = leakyf(v[c], 0.01f); }
    for (int c = 0; c < 4; ++c) {
        float t = ba[c];
        for (int k = 0; k < 4; ++k) t += l[k] * Wa[k * 4 + c];
        float ha = 1.f / (1.f + __expf(-t));
        out[g * 4 + c] = 1.f / (1.f + __expf(-(v[c] * ha)));
    }
}

// ---------------------------------------------------------------------------
// Host orchestration
// ---------------------------------------------------------------------------
extern "C" void kernel_launch(void* const* d_in, const int* in_sizes, int n_in,
                              void* d_out, int out_size, void* d_ws, size_t ws_size,
                              hipStream_t stream) {
    (void)in_sizes; (void)n_in; (void)out_size; (void)ws_size;
    // ---- input mapping (setup_inputs order; params pytree-flattened, sorted keys)
    const float* node_feats = (const float*)d_in[0];   // [100000,1024]
    const float* edge_feats = (const float*)d_in[1];   // [1600000,2]
    const float* affine_W   = (const float*)d_in[2];   // [4,4]
    const float* affine_b   = (const float*)d_in[3];   // [4]
    struct Blk { const float *We, *Wn, *ae, *al, *ar, *bnb, *bng, *bneb, *bneg; };
    Blk blk[5];
    for (int i = 0; i < 5; ++i) {
        int p = 4 + i * 9;   // W_e, W_n, attn_e, attn_l, attn_r, bn_b, bn_g, bne_b, bne_g
        blk[i].We  = (const float*)d_in[p + 0];
        blk[i].Wn  = (const float*)d_in[p + 1];
        blk[i].ae  = (const float*)d_in[p + 2];
        blk[i].al  = (const float*)d_in[p + 3];
        blk[i].ar  = (const float*)d_in[p + 4];
        blk[i].bnb = (const float*)d_in[p + 5];
        blk[i].bng = (const float*)d_in[p + 6];
        blk[i].bneb= (const float*)d_in[p + 7];
        blk[i].bneg= (const float*)d_in[p + 8];
    }
    const float* bn_edge_b  = (const float*)d_in[49];
    const float* bn_edge_g  = (const float*)d_in[50];
    const float* bn_first_b = (const float*)d_in[51];
    const float* bn_first_g = (const float*)d_in[52];
    const float* bn_last_b  = (const float*)d_in[53];
    const float* bn_last_g  = (const float*)d_in[54];
    const float* emb        = (const float*)d_in[55];  // [21,21]
    const float* emb2       = (const float*)d_in[56];  // [11,12]
    const float* f_We = (const float*)d_in[57];
    const float* f_Wn = (const float*)d_in[58];        // [1057,64]
    const float* f_ae = (const float*)d_in[59];
    const float* f_al = (const float*)d_in[60];
    const float* f_ar = (const float*)d_in[61];
    const float* l_We = (const float*)d_in[62];
    const float* l_Wn = (const float*)d_in[63];        // [64,4]
    const float* l_ae = (const float*)d_in[64];
    const float* l_al = (const float*)d_in[65];
    const float* l_ar = (const float*)d_in[66];
    const int* residues  = (const int*)d_in[67];
    const int* secondary = (const int*)d_in[68];
    const int* srcI      = (const int*)d_in[69];
    const int* dstI      = (const int*)d_in[70];
    const int* gids      = (const int*)d_in[71];

    // ---- workspace carve-up
    float* base = (float*)d_ws; size_t off = 0;
    auto alloc = [&](size_t n) { float* p = base + off; off += n; return p; };
    float* ft    = alloc((size_t)NNODES * 64);
    float* hA    = alloc((size_t)NNODES * 64);
    float* hB    = alloc((size_t)NNODES * 64);
    float* eA    = alloc((size_t)NEDGES * 10);
    float* eB    = alloc((size_t)NEDGES * 10);
    float* el    = alloc((size_t)NNODES * 2);
    float* er    = alloc((size_t)NNODES * 2);
    float* ee    = alloc((size_t)NEDGES * 2);
    float* mb    = alloc((size_t)NNODES * 2);
    float* dnm   = alloc((size_t)NNODES * 2);
    float* exb   = alloc((size_t)NEDGES * 2);
    float* ftN   = alloc((size_t)NNODES * 4);
    float* stats = alloc(256);
    float* hg    = alloc(NGRAPH * 4);

    const dim3 B(256);
    auto G = [](long n) { return dim3((unsigned)((n + 255) / 256)); };
    const dim3 GEMMG((NNODES + 127) / 128);

    auto attention = [&](const float* ftp, const float* ein, int Cin,
                         const float* We, const float* ae, const float* al, const float* ar,
                         float* feout, float* outb, int H, int dn, int de) {
        long EH = (long)NEDGES * H;
        k_edge_fe_ee<<<G(NEDGES), B, 0, stream>>>(ein, We, ae, feout, ee, NEDGES, Cin, H, de);
        k_node_el_er<<<G(NNODES), B, 0, stream>>>(ftp, al, ar, el, er, NNODES, H, dn);
        k_fill<<<G((long)NNODES * H), B, 0, stream>>>(mb, (long)NNODES * H, -1e30f);
        k_fill<<<G((long)NNODES * H), B, 0, stream>>>(dnm, (long)NNODES * H, 0.f);
        k_edge_max<<<G(EH), B, 0, stream>>>(el, er, ee, srcI, dstI, mb, EH, H);
        k_edge_expsum<<<G(EH), B, 0, stream>>>(el, er, ee, srcI, dstI, mb, dnm, exb, EH, H);
        k_fill<<<G((long)NNODES * H * dn), B, 0, stream>>>(outb, (long)NNODES * H * dn, 0.f);
        k_edge_scatter<<<G(EH), B, 0, stream>>>(ftp, exb, dnm, srcI, dstI, outb, EH, H, dn);
    };
    auto batchnorm = [&](float* X, long rows, int C, const float* g, const float* b, float slope) {
        k_fill<<<G(2 * C), B, 0, stream>>>(stats, 2 * C, 0.f);
        k_colstats<<<768, B, (size_t)(2 * C * sizeof(float)), stream>>>(X, rows, C, stats);
        k_bn_apply<<<G(rows * C), B, 0, stream>>>(X, rows, C, stats, g, b, slope);
    };

    // ===== first layer: fused-concat WMMA GEMM (K = 1024 dense + 33 fused)
    k_gemm64_wmma<true><<<GEMMG, B, 0, stream>>>(
        node_feats, 1024, f_Wn + 33 * 64, 64, ft, NNODES, 1024,
        f_Wn, emb, emb2, residues, secondary);
    attention(ft, edge_feats, 2, f_We, f_ae, f_al, f_ar, eA, hA, 1, 64, 10);
    batchnorm(hA, NNODES, 64, bn_first_g, bn_first_b, 0.01f);
    batchnorm(eA, NEDGES, 10, bn_edge_g, bn_edge_b, -1.f);

    // ===== 5 residual blocks (K=64 WMMA GEMM, 2 heads x 32)
    float *hin = hA, *hout = hB, *ein = eA, *eout = eB;
    for (int i = 0; i < 5; ++i) {
        k_gemm64_wmma<false><<<GEMMG, B, 0, stream>>>(
            hin, 64, blk[i].Wn, 64, ft, NNODES, 64,
            nullptr, nullptr, nullptr, nullptr, nullptr);
        attention(ft, ein, 10, blk[i].We, blk[i].ae, blk[i].al, blk[i].ar,
                  eout, hout, 2, 32, 5);
        k_add<<<G((long)NNODES * 64), B, 0, stream>>>(hout, hin, (long)NNODES * 64);
        batchnorm(hout, NNODES, 64, blk[i].bng, blk[i].bnb, 0.01f);
        batchnorm(eout, NEDGES, 10, blk[i].bneg, blk[i].bneb, -1.f);
        float* t;
        t = hin; hin = hout; hout = t;
        t = ein; ein = eout; eout = t;
    }

    // ===== last layer (64 -> 4), scalar GEMM
    k_gemm_small<<<G(NNODES), B, 0, stream>>>(hin, l_Wn, ftN, NNODES, 64, 4);
    float* hNC = hout;   // reuse: only N*4 of it
    attention(ftN, ein, 10, l_We, l_ae, l_al, l_ar, eout, hNC, 1, 4, 1);
    batchnorm(hNC, NNODES, 4, bn_last_g, bn_last_b, -1.f);

    // ===== graph pooling + affine-sigmoid head
    k_fill<<<G(NGRAPH * 4), B, 0, stream>>>(hg, NGRAPH * 4, 0.f);
    k_seg_sum<<<G(NNODES), B, 0, stream>>>(hNC, gids, hg, NNODES);
    k_final<<<dim3(1), B, 0, stream>>>(hg, affine_W, affine_b, (float*)d_out);
}
